// Attention_9594956939856
// MI455X (gfx1250) — matrix operand order repaired and checked by reading, not compiled, not run on hardware
//
#include <hip/hip_runtime.h>
#include <hip/hip_bf16.h>

// Problem constants (B, C, H, W) = (8, 256, 64, 64), DK = C/8 = 32
#define B_  8
#define C_  256
#define N_  4096   // H*W
#define DK_ 32

#define CSPLIT   2                  // channel chunks per query tile
#define CCHUNK   (C_ / CSPLIT)      // 128 channels per block
#define CTILES   (CCHUNK / 16)      // 8 WMMA c-tiles per block

typedef __attribute__((ext_vector_type(16))) __bf16       v16bf;
typedef __attribute__((ext_vector_type(8)))  float        v8f;
typedef __attribute__((ext_vector_type(4)))  unsigned int u32x4;

// 16x(bf16) WMMA operand fragment, fillable via integer loads (no scalar __bf16 ops).
union Frag16 {
    v16bf v;
    unsigned int u[8];
    u32x4 q[2];
};

// float -> bf16 (round to nearest even), pure integer math
__device__ __forceinline__ unsigned int f2bf(float f) {
    unsigned int u = __float_as_uint(f);
    unsigned int r = 0x7FFFu + ((u >> 16) & 1u);
    return (u + r) >> 16;   // low 16 bits valid
}
// pack two floats to bf16x2: round-half-up + one v_perm_b32
__device__ __forceinline__ unsigned int pack_bf16(float lo, float hi) {
    unsigned int a = __float_as_uint(lo) + 0x8000u;   // bf16 in bytes [3:2]
    unsigned int b = __float_as_uint(hi) + 0x8000u;
    return __builtin_amdgcn_perm(b, a, 0x07060302u);  // {b[3],b[2],a[3],a[2]}
}

// ---------------------------------------------------------------------------
// Kernel 1: 1x1-conv projections + bf16 conversion.
//   role 0..7 : V channel chunk e = role*32..role*32+31  ->  vb[b][e][n]   (c-major, n contiguous)
//   role 8    : Q                                         ->  qb[b][n][dk] (row per query)
//   role 9    : K                                         ->  kb[b][n][dk] (row per key)
// Weight reads are wave-uniform -> scalar loads; x reads coalesced over n.
// ---------------------------------------------------------------------------
__global__ __launch_bounds__(256)
void proj_kernel(const float* __restrict__ x,
                 const float* __restrict__ Wq, const float* __restrict__ bq,
                 const float* __restrict__ Wk, const float* __restrict__ bk,
                 const float* __restrict__ Wv, const float* __restrict__ bv,
                 unsigned short* __restrict__ qb,
                 unsigned short* __restrict__ kb,
                 unsigned short* __restrict__ vb) {
    const int n    = blockIdx.x * blockDim.x + threadIdx.x;  // pixel within batch
    const int role = blockIdx.y;                             // 0..9
    const int b    = blockIdx.z;

    const float* W;
    const float* bias;
    if (role < 8)       { W = Wv + (size_t)role * 32 * C_; bias = bv + role * 32; }
    else if (role == 8) { W = Wq;                          bias = bq; }
    else                { W = Wk;                          bias = bk; }

    float acc[32];
#pragma unroll
    for (int d = 0; d < 32; ++d) acc[d] = bias[d];

    const float* xp = x + (size_t)b * C_ * N_ + n;
#pragma unroll 4
    for (int c = 0; c < C_; ++c) {
        float xv = xp[(size_t)c * N_];
#pragma unroll
        for (int d = 0; d < 32; ++d)
            acc[d] = fmaf(W[d * C_ + c], xv, acc[d]);
    }

    if (role < 8) {
#pragma unroll
        for (int d = 0; d < 32; ++d)
            vb[((size_t)b * C_ + role * 32 + d) * N_ + n] = (unsigned short)f2bf(acc[d]);
    } else {
        unsigned short* dst = (role == 8 ? qb : kb) + ((size_t)b * N_ + n) * DK_;
#pragma unroll
        for (int d = 0; d < 32; ++d)
            dst[d] = (unsigned short)f2bf(acc[d]);
    }
}

// ---------------------------------------------------------------------------
// Kernel 2: streaming (flash) attention, bf16 WMMA, fp32 accumulate.
// Grid: (qtile, c-chunk, batch). 128 threads = 4 waves; each wave owns 16
// queries x 128 channels. Per 32-key tile:
//   S^T = K_tile(16x32dk) . Q^T(32dk x 16q)   -> 2x v_wmma_f32_16x16x32_bf16
//   online softmax: per-lane over 8 VGPRs + one shfl_xor(16)
//   O^T += V(16c x 32k) . P^T(32k x 16q)      -> 8x wmma into named v8f accums
// Accumulators are EXPLICIT variables (no arrays): every reference is a
// compile-time register -- no scratch spilling. The O-rescale is skipped via
// a ballot-guarded branch whenever no lane's running max changed.
// Epilogue: out = gamma * O/l + x, stored in (b,c,n) layout.
// ---------------------------------------------------------------------------
__global__ __launch_bounds__(128, 1)
void flash_attn_kernel(const unsigned short* __restrict__ qb,
                       const unsigned short* __restrict__ kb,
                       const unsigned short* __restrict__ vb,
                       const float* __restrict__ x,
                       const float* __restrict__ gamma,
                       float* __restrict__ out) {
    const int lane = threadIdx.x & 31;
    const int hl   = lane >> 4;      // half-wave: 0 or 1
    const int l16  = lane & 15;
    const int wave = threadIdx.x >> 5;

    const int b     = blockIdx.z;
    const int cbase = blockIdx.y * CCHUNK;
    const int q0    = blockIdx.x * 64 + wave * 16;

    // Q as B-matrix of S^T=K.Q^T: lane = query col; lanes 0-15 hold dk 0..7,16..23,
    // lanes 16-31 hold dk 8..15,24..31 (same order as the K A-fragments).
    Frag16 qf;
    {
        const unsigned short* qp = qb + ((size_t)b * N_ + (q0 + l16)) * DK_;
        qf.q[0] = *(const u32x4*)(qp + hl * 8);
        qf.q[1] = *(const u32x4*)(qp + 16 + hl * 8);
    }

    // O^T accumulators: 8 named 16x16 fragments (64 VGPRs), never memory-backed.
    v8f o0 = {}, o1 = {}, o2 = {}, o3 = {}, o4 = {}, o5 = {}, o6 = {}, o7 = {};

    float m    = -1e30f;            // running max for query (q0+l16), replicated in both halves
    float lsum = 0.0f;

    const unsigned short* kbase = kb + (size_t)b * N_ * DK_;
    const unsigned short* vbase = vb + ((size_t)b * C_ + cbase) * N_;

#pragma unroll 1
    for (int kt = 0; kt < N_; kt += 32) {
        if (kt + 32 < N_)  // prefetch next K tile row for this lane (global_prefetch_b8)
            __builtin_prefetch(kbase + (size_t)(kt + 32 + l16) * DK_, 0, 1);

        // K A-fragments (M=key, K=dk). 16-bit A layout: lanes 0-15: v0-3 = K0..7,
        // v4-7 = K16..23; lanes 16-31: v0-3 = K8..15, v4-7 = K24..31.
        Frag16 ka0, ka1;
        {
            const unsigned short* kp0 = kbase + (size_t)(kt + l16) * DK_;
            ka0.q[0] = *(const u32x4*)(kp0 + hl * 8);
            ka0.q[1] = *(const u32x4*)(kp0 + 16 + hl * 8);
            const unsigned short* kp1 = kbase + (size_t)(kt + 16 + l16) * DK_;
            ka1.q[0] = *(const u32x4*)(kp1 + hl * 8);
            ka1.q[1] = *(const u32x4*)(kp1 + 16 + hl * 8);
        }
        v8f s0 = {}, s1 = {};
        s0 = __builtin_amdgcn_wmma_f32_16x16x32_bf16(false, ka0.v, false, qf.v,
                                                     (short)0, s0, false, false);
        s1 = __builtin_amdgcn_wmma_f32_16x16x32_bf16(false, ka1.v, false, qf.v,
                                                     (short)0, s1, false, false);
        // s0 VGPR r: (key = kt + r + 8*hl, query = q0+l16); s1: keys kt+16+...

        // ---- online softmax (keys live across 8 VGPRs + partner half-lane)
        float tmax = fmaxf(s0[0], s0[1]);
#pragma unroll
        for (int r = 2; r < 8; ++r) tmax = fmaxf(tmax, s0[r]);
#pragma unroll
        for (int r = 0; r < 8; ++r) tmax = fmaxf(tmax, s1[r]);
        tmax = fmaxf(tmax, __shfl_xor(tmax, 16, 32));

        const float mnew = fmaxf(m, tmax);
        // Rescale only if some lane's running max actually increased; in steady
        // state this skips 64 v_mul + 1 v_exp per tile (uniform s_cbranch).
        if (__ballot(tmax > m)) {
            const float alpha = __expf(m - mnew);   // == 1 for lanes with unchanged max
            lsum *= alpha;
            o0 *= alpha; o1 *= alpha; o2 *= alpha; o3 *= alpha;
            o4 *= alpha; o5 *= alpha; o6 *= alpha; o7 *= alpha;
        }
        m = mnew;

        v8f p0, p1;
        float tsum = 0.0f;
#pragma unroll
        for (int r = 0; r < 8; ++r) { p0[r] = __expf(s0[r] - mnew); tsum += p0[r]; }
#pragma unroll
        for (int r = 0; r < 8; ++r) { p1[r] = __expf(s1[r] - mnew); tsum += p1[r]; }
        tsum += __shfl_xor(tsum, 16, 32);
        lsum += tsum;

        // ---- P^T as B-matrix (K = 32 keys, N = 16 queries), bf16.
        // B order used here (as V): lanes 0-15 VGPR j = keys 2j,2j+1; lanes 16-31 VGPR j = keys 16+2j.
        // Local C/D data covers keys (0..7)+8*hl (p0) and 16+(0..7)+8*hl (p1);
        // one xor-16 shuffle per packed pair completes the fragment.
        unsigned pk0[4], pk1[4];
#pragma unroll
        for (int j = 0; j < 4; ++j) {
            pk0[j] = pack_bf16(p0[2 * j], p0[2 * j + 1]);
            pk1[j] = pack_bf16(p1[2 * j], p1[2 * j + 1]);
        }
        Frag16 pf;
#pragma unroll
        for (int j = 0; j < 4; ++j) {
            unsigned ta = __shfl_xor(pk0[j], 16, 32);   // keys 8+2j  (from half1)
            unsigned tb = __shfl_xor(pk1[j], 16, 32);   // keys 16+2j (from half0)
            pf.u[j]     = hl ? tb     : pk0[j];
            pf.u[j + 4] = hl ? pk1[j] : ta;
        }

        // ---- O^T += V . P^T, one WMMA per named accumulator (V rows contiguous)
#define PV_STEP(CT, OACC)                                                        \
        {                                                                        \
            Frag16 va;                                                           \
            const unsigned short* vp = vbase + (size_t)((CT) * 16 + l16) * N_ + kt; \
            va.q[0] = *(const u32x4*)(vp + hl * 16);                             \
            va.q[1] = *(const u32x4*)(vp + hl * 16 + 8);                         \
            OACC = __builtin_amdgcn_wmma_f32_16x16x32_bf16(false, va.v, false,   \
                       pf.v, (short)0, OACC, false, false);                      \
        }
        PV_STEP(0, o0) PV_STEP(1, o1) PV_STEP(2, o2) PV_STEP(3, o3)
        PV_STEP(4, o4) PV_STEP(5, o5) PV_STEP(6, o6) PV_STEP(7, o7)
#undef PV_STEP
    }

    // ---- epilogue: out[b,c,q] = gamma * O[c,q]/l[q] + x[b,c,q]
    const float g    = gamma[0];
    const float invl = 1.0f / lsum;
    const int   q    = q0 + l16;

#define WRITE_TILE(CT, OACC)                                                     \
    {                                                                            \
        _Pragma("unroll")                                                        \
        for (int r = 0; r < 8; ++r) {                                            \
            const int    c   = cbase + (CT) * 16 + r + 8 * hl;                   \
            const size_t idx = ((size_t)b * C_ + c) * N_ + q;                    \
            out[idx] = g * (OACC[r] * invl) + x[idx];                            \
        }                                                                        \
    }
    WRITE_TILE(0, o0) WRITE_TILE(1, o1) WRITE_TILE(2, o2) WRITE_TILE(3, o3)
    WRITE_TILE(4, o4) WRITE_TILE(5, o5) WRITE_TILE(6, o6) WRITE_TILE(7, o7)
#undef WRITE_TILE
}

// ---------------------------------------------------------------------------
extern "C" void kernel_launch(void* const* d_in, const int* in_sizes, int n_in,
                              void* d_out, int out_size, void* d_ws, size_t ws_size,
                              hipStream_t stream) {
    (void)in_sizes; (void)n_in; (void)out_size; (void)ws_size;
    const float* x     = (const float*)d_in[0];
    const float* Wq    = (const float*)d_in[1];
    const float* bq    = (const float*)d_in[2];
    const float* Wk    = (const float*)d_in[3];
    const float* bk    = (const float*)d_in[4];
    const float* Wv    = (const float*)d_in[5];
    const float* bv    = (const float*)d_in[6];
    const float* gamma = (const float*)d_in[7];
    float*       out   = (float*)d_out;

    // Workspace: qb (2 MB) | kb (2 MB) | vb (16 MB), all bf16
    char* ws = (char*)d_ws;
    const size_t qk_bytes = (size_t)B_ * N_ * DK_ * 2;
    unsigned short* qb = (unsigned short*)(ws);
    unsigned short* kb = (unsigned short*)(ws + qk_bytes);
    unsigned short* vb = (unsigned short*)(ws + 2 * qk_bytes);

    proj_kernel<<<dim3(N_ / 256, 10, B_), 256, 0, stream>>>(
        x, Wq, bq, Wk, bk, Wv, bv, qb, kb, vb);

    flash_attn_kernel<<<dim3(N_ / 64, CSPLIT, B_), 128, 0, stream>>>(
        qb, kb, vb, x, gamma, out);
}
